// IdealDPM_58626303590990
// MI455X (gfx1250) — compile-verified, matrix operand order checked by reading, not run on hardware
//
#include <hip/hip_runtime.h>

// ---------------------------------------------------------------------------
// MI455X (gfx1250) implementation.
// All five matmul stages run on v_wmma_f32_16x16x32_bf16 (bf16 A/B, fp32 acc).
// Attention is a single-pass (flash-style) online softmax: scores never touch
// memory (would be 537 MB otherwise). Operand layouts are chosen so every WMMA
// fragment is two contiguous 16B global/LDS loads per lane (no transposes).
// Softmax row reductions use DPP16 butterflies (VALU-only, no LDS round trips).
// ---------------------------------------------------------------------------

#define Bc  32
#define Tc  1024
#define Ec  256
#define Hc  4
#define HDc 64
#define Pc  32
#define BT  (Bc * Tc)

typedef __attribute__((ext_vector_type(16))) __bf16 v16bf;
typedef __attribute__((ext_vector_type(8)))  __bf16 v8bf;
typedef __attribute__((ext_vector_type(8)))  float  v8f;

__device__ inline __bf16 f2bf(float f) {
  union { float f; unsigned u; } a; a.f = f;
  unsigned r = a.u + 0x7FFFu + ((a.u >> 16) & 1u);   // round-to-nearest-even
  unsigned short h = (unsigned short)(r >> 16);
  __bf16 o; __builtin_memcpy(&o, &h, 2); return o;
}

// A/B fragment loader for wmma_f32_16x16x32_bf16.
// Per the CDNA5 ISA layout: lane l holds row/col (l&15); K chunks
// [kb, kb+8) and [kb+16, kb+24) with kb = (l<16) ? 0 : 8.
// Caller passes p already offset by kb; we read p[0..7] and p[16..23].
__device__ inline v16bf load16(const __bf16* p) {
  v8bf lo = *(const v8bf*)p;
  v8bf hi = *(const v8bf*)(p + 16);
  v16bf o;
#pragma unroll
  for (int i = 0; i < 8; i++) { o[i] = lo[i]; o[i + 8] = hi[i]; }
  return o;
}

__device__ inline v8f wmma_bf(v16bf a, v16bf b, v8f c) {
  return __builtin_amdgcn_wmma_f32_16x16x32_bf16(
      /*neg_a=*/false, a, /*neg_b=*/false, b,
      /*c_mod=*/(short)0, c, /*reuse_a=*/false, /*reuse_b=*/false);
}

// ---- DPP16 lane permutes (within each 16-lane row; EXEC is all-1s here) ----
template <int CTRL>
__device__ inline float dpp_perm(float x) {
  int xi = __builtin_bit_cast(int, x);
  int r  = __builtin_amdgcn_update_dpp(0, xi, CTRL, 0xF, 0xF, true);
  return __builtin_bit_cast(float, r);
}
// Butterfly over a 16-lane row with involutions {xor1, xor2, xor7, xor15}:
// after quad_perm steps each lane holds its quad q; row_half_mirror (lane^7)
// merges q with q^4; row_mirror (lane^15) merges q^8 and q^12 -> all 16 lanes.
__device__ inline float redmax16(float v) {
  v = fmaxf(v, dpp_perm<0xB1>(v));    // quad_perm(1,0,3,2)  : lane ^ 1
  v = fmaxf(v, dpp_perm<0x4E>(v));    // quad_perm(2,3,0,1)  : lane ^ 2
  v = fmaxf(v, dpp_perm<0x141>(v));   // row_half_mirror     : lane ^ 7
  v = fmaxf(v, dpp_perm<0x140>(v));   // row_mirror          : lane ^ 15
  return v;
}
__device__ inline float redsum16(float v) {
  v += dpp_perm<0xB1>(v);
  v += dpp_perm<0x4E>(v);
  v += dpp_perm<0x141>(v);
  v += dpp_perm<0x140>(v);
  return v;
}

// ---------------------------------------------------------------------------
// fp32 -> bf16 convert
__global__ void k_f2bf(const float* __restrict__ in, __bf16* __restrict__ out, int n) {
  int i = blockIdx.x * blockDim.x + threadIdx.x;
  if (i < n) out[i] = f2bf(in[i]);
}

// ---------------------------------------------------------------------------
// Generic WMMA GEMM:  out[M,N] = A_bf16[M,K] @ W_bf16[N,K]^T + bias[N]
// One wave computes one 16x16 tile. EPI: 0 = bf16 row-major out,
// 1 = f32 row-major out, 2 = scatter into q/k (per-head) and v (transposed).
template <int EPI>
__global__ void k_gemm(const __bf16* __restrict__ A, const __bf16* __restrict__ W,
                       const float* __restrict__ bias,
                       float* __restrict__ outF, __bf16* __restrict__ outB,
                       __bf16* __restrict__ qb, __bf16* __restrict__ kb_,
                       __bf16* __restrict__ vT,
                       int M, int N, int K, int tilesN) {
  int wave = (int)((blockIdx.x * blockDim.x + threadIdx.x) >> 5);
  int lane = threadIdx.x & 31;
  int tm = wave / tilesN, tn = wave % tilesN;
  if (tm * 16 >= M) return;
  int r  = lane & 15;
  int hi = lane >> 4;
  int kb = hi ? 8 : 0;

  const __bf16* arow = A + (size_t)(tm * 16 + r) * K;
  const __bf16* wrow = W + (size_t)(tn * 16 + r) * K;

  v8f acc = {};
  for (int k0 = 0; k0 < K; k0 += 32) {
    v16bf a = load16(arow + k0 + kb);
    v16bf b = load16(wrow + k0 + kb);
    acc = wmma_bf(a, b, acc);
  }
  float bv = bias ? bias[tn * 16 + r] : 0.0f;

#pragma unroll
  for (int rr = 0; rr < 8; rr++) {
    int grow = tm * 16 + rr + hi * 8;     // C layout: M = vgpr + 8*half
    int gcol = tn * 16 + r;               // N = lane & 15
    float v = acc[rr] + bv;
    if (EPI == 0) {
      outB[(size_t)grow * N + gcol] = f2bf(v);
    } else if (EPI == 1) {
      outF[(size_t)grow * N + gcol] = v;
    } else {
      // qkv scatter: grow = token (b*T + t), gcol in [0, 768)
      int b = grow / Tc, t = grow % Tc;
      int sec = gcol >> 8;                // 0=q 1=k 2=v
      int e = gcol & 255;
      int hh = e >> 6, d = e & 63;
      __bf16 bvv = f2bf(v);
      size_t bh = (size_t)b * Hc + hh;
      if (sec == 0)      qb [(bh * Tc + t) * HDc + d] = bvv;
      else if (sec == 1) kb_[(bh * Tc + t) * HDc + d] = bvv;
      else               vT [(bh * HDc + d) * Tc + t] = bvv;   // transposed
    }
  }
}

// ---------------------------------------------------------------------------
// Flash attention: 1 wave owns one (b,h, 16-query tile); streams 32-key blocks.
// q,k: [B*H, T, HD] bf16.  vT: [B*H, HD, T] bf16.  ctx: [B*T, E] bf16.
__global__ void k_attn(const __bf16* __restrict__ q, const __bf16* __restrict__ k,
                       const __bf16* __restrict__ vT, __bf16* __restrict__ ctx) {
  __shared__ __align__(16) __bf16 psh[8][16 * 32];   // per-wave P tile
  int lane = threadIdx.x & 31;
  int wv   = threadIdx.x >> 5;
  int gw   = blockIdx.x * 8 + wv;
  if (gw >= (Bc * Hc) * (Tc / 16)) return;
  int bh = gw >> 6;          // 64 q-tiles per (b,h)
  int qt = gw & 63;
  int r  = lane & 15;
  int hi = lane >> 4;
  int kb = hi ? 8 : 0;

  // Q fragments for K-offsets 0 and 32 (HD = 64), reused for the whole row.
  const __bf16* qrow = q + ((size_t)bh * Tc + qt * 16 + r) * HDc;
  v16bf qa0 = load16(qrow + 0  + kb);
  v16bf qa1 = load16(qrow + 32 + kb);

  const __bf16* kbase = k  + (size_t)bh * Tc * HDc;
  const __bf16* vbase = vT + (size_t)bh * HDc * Tc;

  v8f acc[4]; float mrun[8], lrun[8];
#pragma unroll
  for (int c = 0; c < 4; c++) acc[c] = (v8f){};
#pragma unroll
  for (int rr = 0; rr < 8; rr++) { mrun[rr] = -1e30f; lrun[rr] = 0.0f; }

  for (int kc = 0; kc < Tc; kc += 32) {
    // prefetch next key block (K rows + V rows) into cache
    if (kc + 32 < Tc) {
      __builtin_prefetch(kbase + (size_t)(kc + 32 + r) * HDc, 0, 3);
      __builtin_prefetch(vbase + (size_t)r * Tc + kc + 32, 0, 3);
    }

    // ---- S = (Q @ K^T) * 1/sqrt(64), two 16x16 tiles (keys kc..+15, +16..+31)
    const __bf16* kr0 = kbase + (size_t)(kc + r) * HDc;
    const __bf16* kr1 = kbase + (size_t)(kc + 16 + r) * HDc;
    v8f s0 = {}, s1 = {};
    s0 = wmma_bf(qa0, load16(kr0 + 0  + kb), s0);
    s0 = wmma_bf(qa1, load16(kr0 + 32 + kb), s0);
    s1 = wmma_bf(qa0, load16(kr1 + 0  + kb), s1);
    s1 = wmma_bf(qa1, load16(kr1 + 32 + kb), s1);

    // ---- online softmax across this 32-key block (DPP16 row reductions)
    float p0[8], p1[8];
#pragma unroll
    for (int rr = 0; rr < 8; rr++) {
      float a = s0[rr] * 0.125f, b = s1[rr] * 0.125f;
      float vm = redmax16(fmaxf(a, b));            // row max over 16 lanes
      float mnew  = fmaxf(mrun[rr], vm);
      float alpha = __expf(mrun[rr] - mnew);
      float e0 = __expf(a - mnew), e1 = __expf(b - mnew);
      float ts = redsum16(e0 + e1);                // row sum over 16 lanes
      lrun[rr] = lrun[rr] * alpha + ts;
      mrun[rr] = mnew;
#pragma unroll
      for (int c = 0; c < 4; c++) acc[c][rr] *= alpha;
      p0[rr] = e0; p1[rr] = e1;
    }

    // ---- stage P (C-layout) into LDS row-major, reload as A fragment
#pragma unroll
    for (int rr = 0; rr < 8; rr++) {
      psh[wv][(rr + hi * 8) * 32 + r]      = f2bf(p0[rr]);
      psh[wv][(rr + hi * 8) * 32 + 16 + r] = f2bf(p1[rr]);
    }
    v16bf pa = load16(&psh[wv][r * 32 + kb]);    // same-wave LDS RAW: dscnt wait

    // ---- acc += P[16,32] @ V[32,64]  (vT rows are key-contiguous)
#pragma unroll
    for (int c = 0; c < 4; c++) {
      const __bf16* vrow = vbase + (size_t)(c * 16 + r) * Tc + kc + kb;
      acc[c] = wmma_bf(pa, load16(vrow), acc[c]);
    }
  }

  // ---- normalize and write ctx[token, h*64 + d] as bf16
  int b = bh >> 2, h = bh & 3;
#pragma unroll
  for (int c = 0; c < 4; c++) {
#pragma unroll
    for (int rr = 0; rr < 8; rr++) {
      int token = b * Tc + qt * 16 + rr + hi * 8;
      ctx[(size_t)token * Ec + h * HDc + c * 16 + r] = f2bf(acc[c][rr] / lrun[rr]);
    }
  }
}

// ---------------------------------------------------------------------------
// boundary MLP: sigmoid(w2 . relu(W1 @ ao + b1) + b2), one thread per token
__global__ void k_bscore(const float* __restrict__ ao, const float* __restrict__ w1,
                         const float* __restrict__ b1, const float* __restrict__ w2,
                         const float* __restrict__ b2, float* __restrict__ bs) {
  int tok = blockIdx.x * blockDim.x + threadIdx.x;
  if (tok >= BT) return;
  const float* row = ao + (size_t)tok * Ec;
  float s2 = b2[0];
  for (int j = 0; j < 32; j++) {
    const float* w = w1 + j * Ec;
    float a = b1[j];
    for (int i = 0; i < Ec; i++) a += row[i] * w[i];
    s2 += w2[j] * fmaxf(a, 0.0f);
  }
  bs[tok] = 1.0f / (1.0f + __expf(-s2));
}

// inclusive scan over T per batch -> normalized cum boundary -> patch id
__global__ void k_scan_pid(const float* __restrict__ bs, int* __restrict__ pid) {
  __shared__ float s[Tc];
  int b = blockIdx.x, t = threadIdx.x;
  s[t] = bs[b * Tc + t];
  __syncthreads();
  for (int off = 1; off < Tc; off <<= 1) {
    float add = (t >= off) ? s[t - off] : 0.0f;
    __syncthreads();
    s[t] += add;
    __syncthreads();
  }
  float cbp = s[t] / (s[Tc - 1] + 1e-6f);
  int p = (int)floorf(cbp * (float)Pc);
  pid[b * Tc + t] = p < (Pc - 1) ? (p < 0 ? 0 : p) : (Pc - 1);
}

__global__ void k_zero(float* p, int n) {
  int i = blockIdx.x * blockDim.x + threadIdx.x;
  if (i < n) p[i] = 0.0f;
}

// scatter-add pooling: one block per token, one thread per feature
__global__ void k_pool(const float* __restrict__ ao, const int* __restrict__ pid,
                       float* __restrict__ pe, float* __restrict__ cnt) {
  int tok = blockIdx.x;
  int e = threadIdx.x;
  int b = tok / Tc;
  int p = pid[tok];
  atomicAdd(&pe[((size_t)b * Pc + p) * Ec + e], ao[(size_t)tok * Ec + e]);
  if (e == 0) atomicAdd(&cnt[b * Pc + p], 1.0f);
}

__global__ void k_div_bf(const float* __restrict__ pe, const float* __restrict__ cnt,
                         __bf16* __restrict__ pe_bf) {
  int row = blockIdx.x, e = threadIdx.x;
  float c = fmaxf(cnt[row], 1.0f);
  pe_bf[(size_t)row * Ec + e] = f2bf(pe[(size_t)row * Ec + e] / c);
}

// ---------------------------------------------------------------------------
extern "C" void kernel_launch(void* const* d_in, const int* in_sizes, int n_in,
                              void* d_out, int out_size, void* d_ws, size_t ws_size,
                              hipStream_t stream) {
  const float* x      = (const float*)d_in[0];
  const float* ip_w   = (const float*)d_in[1];
  const float* ip_b   = (const float*)d_in[2];
  const float* inp_w  = (const float*)d_in[3];
  const float* inp_b  = (const float*)d_in[4];
  const float* out_w  = (const float*)d_in[5];
  const float* out_b  = (const float*)d_in[6];
  const float* bp_w1  = (const float*)d_in[7];
  const float* bp_b1  = (const float*)d_in[8];
  const float* bp_w2  = (const float*)d_in[9];
  const float* bp_b2  = (const float*)d_in[10];
  const float* pr_w   = (const float*)d_in[11];
  const float* pr_b   = (const float*)d_in[12];

  // ---- workspace carve (256B aligned)
  char* w = (char*)d_ws;
  auto carve = [&](size_t bytes) { void* p = (void*)w; w += (bytes + 255) & ~(size_t)255; return p; };
  __bf16* x_bf   = (__bf16*)carve((size_t)BT * 32 * 2);
  __bf16* h_bf   = (__bf16*)carve((size_t)BT * Ec * 2);
  __bf16* q_bf   = (__bf16*)carve((size_t)BT * Ec * 2);
  __bf16* k_bf   = (__bf16*)carve((size_t)BT * Ec * 2);
  __bf16* vT_bf  = (__bf16*)carve((size_t)BT * Ec * 2);
  __bf16* ctx_bf = (__bf16*)carve((size_t)BT * Ec * 2);
  float*  ao     = (float*) carve((size_t)BT * Ec * 4);
  float*  bs     = (float*) carve((size_t)BT * 4);
  int*    pid    = (int*)   carve((size_t)BT * 4);
  float*  pe     = (float*) carve((size_t)Bc * Pc * Ec * 4);   // pe then cnt contiguous
  float*  cnt    = (float*) carve((size_t)Bc * Pc * 4);
  __bf16* pe_bf  = (__bf16*)carve((size_t)Bc * Pc * Ec * 2);
  __bf16* wip    = (__bf16*)carve((size_t)Ec * 32 * 2);
  __bf16* winp   = (__bf16*)carve((size_t)3 * Ec * Ec * 2);
  __bf16* wout   = (__bf16*)carve((size_t)Ec * Ec * 2);
  __bf16* wpr    = (__bf16*)carve((size_t)Ec * Ec * 2);

  // ---- converts
  k_f2bf<<<(BT * 32 + 255) / 256, 256, 0, stream>>>(x, x_bf, BT * 32);
  k_f2bf<<<(Ec * 32 + 255) / 256, 256, 0, stream>>>(ip_w, wip, Ec * 32);
  k_f2bf<<<(3 * Ec * Ec + 255) / 256, 256, 0, stream>>>(inp_w, winp, 3 * Ec * Ec);
  k_f2bf<<<(Ec * Ec + 255) / 256, 256, 0, stream>>>(out_w, wout, Ec * Ec);
  k_f2bf<<<(Ec * Ec + 255) / 256, 256, 0, stream>>>(pr_w, wpr, Ec * Ec);

  // ---- ip projection: h[BT,256] = x[BT,32] @ ip_w^T + b   (K=32 -> 1 WMMA/tile)
  {
    int tilesN = Ec / 16, waves = (BT / 16) * tilesN;
    k_gemm<0><<<waves / 8, 256, 0, stream>>>(x_bf, wip, ip_b, nullptr, h_bf,
                                             nullptr, nullptr, nullptr, BT, Ec, 32, tilesN);
  }
  // ---- qkv projection with per-head scatter (v stored transposed)
  {
    int tilesN = 3 * Ec / 16, waves = (BT / 16) * tilesN;
    k_gemm<2><<<waves / 8, 256, 0, stream>>>(h_bf, winp, inp_b, nullptr, nullptr,
                                             q_bf, k_bf, vT_bf, BT, 3 * Ec, Ec, tilesN);
  }
  // ---- flash attention
  k_attn<<<(Bc * Hc * (Tc / 16)) / 8, 256, 0, stream>>>(q_bf, k_bf, vT_bf, ctx_bf);

  // ---- out projection (fp32 out, feeds MLP + pooling)
  {
    int tilesN = Ec / 16, waves = (BT / 16) * tilesN;
    k_gemm<1><<<waves / 8, 256, 0, stream>>>(ctx_bf, wout, out_b, ao, nullptr,
                                             nullptr, nullptr, nullptr, BT, Ec, Ec, tilesN);
  }
  // ---- boundary scores, scan -> patch ids
  k_bscore<<<BT / 256, 256, 0, stream>>>(ao, bp_w1, bp_b1, bp_w2, bp_b2, bs);
  k_scan_pid<<<Bc, Tc, 0, stream>>>(bs, pid);

  // ---- segment-mean pooling (pe and cnt are contiguous: one zero pass)
  {
    int nz = Bc * Pc * Ec + Bc * Pc;
    k_zero<<<(nz + 255) / 256, 256, 0, stream>>>(pe, nz);
  }
  k_pool<<<BT, Ec, 0, stream>>>(ao, pid, pe, cnt);
  k_div_bf<<<Bc * Pc, Ec, 0, stream>>>(pe, cnt, pe_bf);

  // ---- final projection -> d_out [B,P,E] fp32
  {
    int M = Bc * Pc, tilesN = Ec / 16, waves = (M / 16) * tilesN;
    k_gemm<1><<<waves / 8, 256, 0, stream>>>(pe_bf, wpr, pr_b, (float*)d_out, nullptr,
                                             nullptr, nullptr, nullptr, M, Ec, Ec, tilesN);
  }
}
